// EnhancedEdgeAwareGNN_40303973105673
// MI455X (gfx1250) — matrix operand order repaired and checked by reading, not compiled, not run on hardware
//
#include <hip/hip_runtime.h>
#include <hip/hip_bf16.h>

typedef __attribute__((ext_vector_type(16))) _Float16 v16h;
typedef __attribute__((ext_vector_type(8)))  _Float16 v8h;
typedef __attribute__((ext_vector_type(8)))  float    v8f;

#define CDIV(a, b) (((a) + (b) - 1) / (b))

// ---------------- ordered-float encoding for atomicMax-based segment max ----
__device__ __forceinline__ unsigned fenc(float f) {
    unsigned u = __float_as_uint(f);
    return (u & 0x80000000u) ? ~u : (u | 0x80000000u);
}
__device__ __forceinline__ float fdec(unsigned e) {
    if (e == 0u) return 0.0f;  // untouched slot == no incoming edges -> amax := 0
    return (e & 0x80000000u) ? __uint_as_float(e & 0x7fffffffu)
                             : __uint_as_float(~e);
}
__device__ __forceinline__ float sigmoidf_(float x) { return 1.0f / (1.0f + __expf(-x)); }

// ---------------- f32 -> f16 convert (grid-stride) --------------------------
__global__ void cvt_f32_f16(const float* __restrict__ in, _Float16* __restrict__ out, size_t n) {
    size_t i = (size_t)blockIdx.x * blockDim.x + threadIdx.x;
    size_t stride = (size_t)gridDim.x * blockDim.x;
    for (; i < n; i += stride) out[i] = (_Float16)in[i];
}

// ---------------- pack f32 weights [K,Nc] row-major into WMMA B fragments ---
// fragment (kt,nt): lane l<16 holds col nt*16+l, K = kt*32 + 0..7, 16..23 (elems 0..15)
//                   lane l>=16 holds same col,  K = kt*32 + 8..15, 24..31
__global__ void pack_b_f16(const float* __restrict__ W, _Float16* __restrict__ Bp,
                           int K, int Nc) {
    int idx = blockIdx.x * blockDim.x + threadIdx.x;
    int total = K * Nc;
    if (idx >= total) return;
    int elem = idx & 15;
    int lane = (idx >> 4) & 31;
    int frag = idx >> 9;
    int ntTiles = Nc >> 4;
    int nt = frag % ntTiles;
    int kt = frag / ntTiles;
    int n = nt * 16 + (lane & 15);
    int k = kt * 32 + ((lane >> 4) << 3) + (elem & 7) + ((elem >> 3) << 4);
    Bp[idx] = (_Float16)W[(size_t)k * Nc + n];
}

// ---------------- WMMA GEMM: C[M,Nc] = act(A[M,K]f16 @ B + bias) ------------
// one wave per 16-row tile, 4 n-tiles (64 cols) per wave; K multiple of 32.
__global__ void gemm_wmma_f16(const _Float16* __restrict__ A,
                              const _Float16* __restrict__ Bp,
                              float* __restrict__ C,
                              const float* __restrict__ bias,
                              int M, int K, int Nc, int relu) {
    const int lane = threadIdx.x & 31;
    const int wave = threadIdx.x >> 5;
    const int mtile = blockIdx.x * (blockDim.x >> 5) + wave;
    if (mtile * 16 >= M) return;                    // wave-uniform: EXEC stays all-1s
    const int ntTiles = Nc >> 4;
    const int nt0 = blockIdx.y * 4;
    const int row = mtile * 16 + (lane & 15);
    const int kphase = (lane >> 4) << 3;            // 0 for lanes 0-15, 8 for 16-31
    v8f acc[4] = {};
    for (int k0 = 0; k0 < K; k0 += 32) {
        const _Float16* ap = A + (size_t)row * K + k0 + kphase;
        __builtin_prefetch(ap + 32, 0, 1);
        v8h lo = *(const v8h*)ap;                   // K = k0+kphase+0..7
        v8h hi = *(const v8h*)(ap + 16);            // K = k0+kphase+16..23
        v16h af;
#pragma unroll
        for (int i = 0; i < 8; ++i) { af[i] = lo[i]; af[i + 8] = hi[i]; }
        const int kt = k0 >> 5;
        const _Float16* bb = Bp + (((size_t)kt * ntTiles + nt0) * 32 + lane) * 16;
#pragma unroll
        for (int j = 0; j < 4; ++j) {
            v16h bf = *(const v16h*)(bb + (size_t)j * 32 * 16);
            acc[j] = __builtin_amdgcn_wmma_f32_16x16x32_f16(
                false, af, false, bf, (short)0, acc[j], false, false);
        }
    }
    const int mrow = mtile * 16 + ((lane >> 4) << 3);   // VGPR r -> M = r (+8 hi lanes)
    const int ncol = lane & 15;
#pragma unroll
    for (int j = 0; j < 4; ++j) {
        const int n = (nt0 + j) * 16 + ncol;
        const float bv = bias ? bias[n] : 0.0f;
#pragma unroll
        for (int r = 0; r < 8; ++r) {
            float v = acc[j][r] + bv;
            if (relu) v = v > 0.0f ? v : 0.0f;
            C[(size_t)(mrow + r) * Nc + n] = v;
        }
    }
}

// ---------------- node / edge embeddings ------------------------------------
__global__ void node_embed(const float* __restrict__ x, const float* __restrict__ W,
                           const float* __restrict__ b, float* __restrict__ xh) {
    __shared__ float xl[8];
    int n = blockIdx.x, c = threadIdx.x;
    if (c < 8) xl[c] = x[(size_t)n * 8 + c];
    __syncthreads();
    float acc = b[c];
#pragma unroll
    for (int k = 0; k < 8; ++k) acc += xl[k] * W[k * 128 + c];
    xh[(size_t)n * 128 + c] = acc;
}

__global__ void edge_embed(const float* __restrict__ ea, const float* __restrict__ W,
                           const float* __restrict__ b, float* __restrict__ out) {
    __shared__ float el[4];
    int e = blockIdx.x, c = threadIdx.x;
    if (c < 4) el[c] = ea[(size_t)e * 4 + c];
    __syncthreads();
    float acc = b[c];
#pragma unroll
    for (int k = 0; k < 4; ++k) acc += el[k] * W[k * 128 + c];
    out[(size_t)e * 128 + c] = acc;
}

// bias1eff[c] = attn_b1[c] + sum_k vemb[k] * attn_w1[128+k, c], vemb = vnf@w_vnf+b
__global__ void make_bias1(const float* __restrict__ vnf, const float* __restrict__ w_vnf,
                           const float* __restrict__ b_vnf, const float* __restrict__ attn_w1,
                           const float* __restrict__ attn_b1, float* __restrict__ bias1) {
    __shared__ float vemb[128];
    int c = threadIdx.x;
    float s = b_vnf[c];
#pragma unroll
    for (int k = 0; k < 6; ++k) s += vnf[k] * w_vnf[k * 128 + c];
    vemb[c] = s;
    __syncthreads();
    float o = attn_b1[c];
    for (int k = 0; k < 128; ++k) o += vemb[k] * attn_w1[(size_t)(128 + k) * 128 + c];
    bias1[c] = o;
}

// sigmoid(h2 . w3 + b3), then e_enh = ef * att ; one wave per edge
__global__ void edge_att(const float* __restrict__ h2, const float* __restrict__ w3,
                         const float* __restrict__ b3, const float* __restrict__ ef,
                         float* __restrict__ e_enh, float* __restrict__ att, int E) {
    int gid = blockIdx.x * blockDim.x + threadIdx.x;
    int e = gid >> 5, lane = gid & 31;
    if (e >= E) return;
    const float* row = h2 + (size_t)e * 64;
    float s = row[lane] * w3[lane] + row[lane + 32] * w3[lane + 32];
#pragma unroll
    for (int off = 16; off; off >>= 1) s += __shfl_xor(s, off, 32);
    float a = sigmoidf_(s + b3[0]);
    if (lane == 0) att[e] = a;
    const float* efr = ef + (size_t)e * 128;
    float* er = e_enh + (size_t)e * 128;
#pragma unroll
    for (int r = 0; r < 4; ++r) er[r * 32 + lane] = efr[r * 32 + lane] * a;
}

// W_eff[k,h] = sum_c gat_we[k, h*128+c] * att_e[h,c]   (folds the E x 128 x 512 GEMM)
__global__ void make_weff(const float* __restrict__ gwe, const float* __restrict__ atte,
                          float* __restrict__ weff) {
    int idx = blockIdx.x * blockDim.x + threadIdx.x;
    if (idx >= 512) return;
    int k = idx >> 2, h = idx & 3;
    float s = 0.0f;
    for (int c = 0; c < 128; ++c) s += gwe[(size_t)k * 512 + h * 128 + c] * atte[h * 128 + c];
    weff[k * 4 + h] = s;
}

// escore[e,h] = e_enh[e,:] @ W_eff[:,h] ; one wave per edge
__global__ void edge_score(const float* __restrict__ e_enh, const float* __restrict__ weff,
                           float* __restrict__ escore, int E) {
    int gid = blockIdx.x * blockDim.x + threadIdx.x;
    int e = gid >> 5, lane = gid & 31;
    if (e >= E) return;
    const float* row = e_enh + (size_t)e * 128;
    float s0 = 0, s1 = 0, s2 = 0, s3 = 0;
#pragma unroll
    for (int r = 0; r < 4; ++r) {
        int k = r * 32 + lane;
        float v = row[k];
        s0 += v * weff[k * 4 + 0]; s1 += v * weff[k * 4 + 1];
        s2 += v * weff[k * 4 + 2]; s3 += v * weff[k * 4 + 3];
    }
#pragma unroll
    for (int off = 16; off; off >>= 1) {
        s0 += __shfl_xor(s0, off, 32); s1 += __shfl_xor(s1, off, 32);
        s2 += __shfl_xor(s2, off, 32); s3 += __shfl_xor(s3, off, 32);
    }
    if (lane == 0) {
        escore[(size_t)e * 4 + 0] = s0; escore[(size_t)e * 4 + 1] = s1;
        escore[(size_t)e * 4 + 2] = s2; escore[(size_t)e * 4 + 3] = s3;
    }
}

// a_src[n,h]/a_dst[n,h] from xs[n,512] ; one wave per node
__global__ void node_scores(const float* __restrict__ xs, const float* __restrict__ Wsrc,
                            const float* __restrict__ Wdst, float* __restrict__ asrc,
                            float* __restrict__ adst, int N) {
    int gid = blockIdx.x * blockDim.x + threadIdx.x;
    int n = gid >> 5, lane = gid & 31;
    if (n >= N) return;
    const float* row = xs + (size_t)n * 512;
#pragma unroll
    for (int h = 0; h < 4; ++h) {
        float s = 0, t = 0;
#pragma unroll
        for (int r = 0; r < 4; ++r) {
            int c = r * 32 + lane;
            float v = row[h * 128 + c];
            s += v * Wsrc[h * 128 + c];
            t += v * Wdst[h * 128 + c];
        }
#pragma unroll
        for (int off = 16; off; off >>= 1) {
            s += __shfl_xor(s, off, 32); t += __shfl_xor(t, off, 32);
        }
        if (lane == 0) { asrc[(size_t)n * 4 + h] = s; adst[(size_t)n * 4 + h] = t; }
    }
}

__global__ void edge_logits(const int* __restrict__ src, const int* __restrict__ dst,
                            const float* __restrict__ asrc, const float* __restrict__ adst,
                            const float* __restrict__ escore, float* __restrict__ a,
                            unsigned* __restrict__ amax, int E4) {
    int i = blockIdx.x * blockDim.x + threadIdx.x;
    if (i >= E4) return;
    int e = i >> 2, h = i & 3;
    int d = dst[e];
    float v = asrc[(size_t)src[e] * 4 + h] + adst[(size_t)d * 4 + h] + escore[i];
    v = v > 0.0f ? v : 0.2f * v;  // leaky relu
    a[i] = v;
    atomicMax(&amax[(size_t)d * 4 + h], fenc(v));
}

__global__ void edge_exp(const int* __restrict__ dst, float* __restrict__ a,
                         const unsigned* __restrict__ amax, float* __restrict__ denom, int E4) {
    int i = blockIdx.x * blockDim.x + threadIdx.x;
    if (i >= E4) return;
    int e = i >> 2, h = i & 3;
    int d = dst[e];
    float ae = __expf(a[i] - fdec(amax[(size_t)d * 4 + h]));
    a[i] = ae;
    atomicAdd(&denom[(size_t)d * 4 + h], ae);
}

// agg[dst, h*128+c] += xs[src, h*128+c] * alpha[e,h] ; one 128-thread block per edge
__global__ void edge_scatter(const int* __restrict__ src, const int* __restrict__ dst,
                             const float* __restrict__ ae, const float* __restrict__ denom,
                             const float* __restrict__ xs, float* __restrict__ agg) {
    int e = blockIdx.x, c = threadIdx.x;
    int s = src[e], d = dst[e];
    __shared__ float alph[4];
    if (c < 4) alph[c] = ae[(size_t)e * 4 + c] / (denom[(size_t)d * 4 + c] + 1e-16f);
    __syncthreads();
    const float* xrow = xs + (size_t)s * 512;
    float* arow = agg + (size_t)d * 512;
#pragma unroll
    for (int h = 0; h < 4; ++h) atomicAdd(&arow[h * 128 + c], xrow[h * 128 + c] * alph[h]);
}

// head-mean + bias + layernorm + residual ; 128-thread block per node
__global__ void node_update(const float* __restrict__ agg, const float* __restrict__ gb,
                            const float* __restrict__ lng, const float* __restrict__ lnb,
                            const float* __restrict__ xh_in, float* __restrict__ xh_out) {
    __shared__ float red[4];
    int n = blockIdx.x, c = threadIdx.x;
    const float* arow = agg + (size_t)n * 512;
    float v = 0.25f * (arow[c] + arow[128 + c] + arow[256 + c] + arow[384 + c]) + gb[c];
    float s = v;
#pragma unroll
    for (int off = 16; off; off >>= 1) s += __shfl_xor(s, off, 32);
    if ((c & 31) == 0) red[c >> 5] = s;
    __syncthreads();
    float mu = (red[0] + red[1] + red[2] + red[3]) * (1.0f / 128.0f);
    __syncthreads();
    float d = v - mu, dv = d * d;
#pragma unroll
    for (int off = 16; off; off >>= 1) dv += __shfl_xor(dv, off, 32);
    if ((c & 31) == 0) red[c >> 5] = dv;
    __syncthreads();
    float var = (red[0] + red[1] + red[2] + red[3]) * (1.0f / 128.0f);
    float hn = d * rsqrtf(var + 1e-5f) * lng[c] + lnb[c];
    xh_out[(size_t)n * 128 + c] = hn + xh_in[(size_t)n * 128 + c];
}

// ---------------- tail: vnf encoder, Set2Set, head MLP ----------------------
__global__ void venc_kernel(const float* __restrict__ vnf, const float* __restrict__ w1,
                            const float* __restrict__ b1, const float* __restrict__ w2,
                            const float* __restrict__ b2, float* __restrict__ v) {
    __shared__ float t[128];
    int c = threadIdx.x;
    float s = b1[c];
#pragma unroll
    for (int k = 0; k < 6; ++k) s += vnf[k] * w1[k * 128 + c];
    t[c] = s > 0.0f ? s : 0.0f;
    __syncthreads();
    float o = b2[c];
    for (int k = 0; k < 128; ++k) o += t[k] * w2[k * 128 + c];
    v[c] = o;
}

__global__ void lstm_step(const float* __restrict__ wih, const float* __restrict__ whh,
                          const float* __restrict__ bih, const float* __restrict__ bhh,
                          float* __restrict__ qstar, float* __restrict__ hS,
                          float* __restrict__ cS) {
    __shared__ float g[512];
    __shared__ float q[256];
    __shared__ float hl[128];
    int t = threadIdx.x;
    if (t < 256) q[t] = qstar[t];
    if (t < 128) hl[t] = hS[t];
    __syncthreads();
    float s = bih[t] + bhh[t];
    for (int k = 0; k < 256; ++k) s += wih[(size_t)t * 256 + k] * q[k];
    for (int k = 0; k < 128; ++k) s += whh[(size_t)t * 128 + k] * hl[k];
    g[t] = s;
    __syncthreads();
    if (t < 128) {
        float gi = g[t], gf = g[128 + t], gg = g[256 + t], go = g[384 + t];
        float c2 = sigmoidf_(gf) * cS[t] + sigmoidf_(gi) * tanhf(gg);
        cS[t] = c2;
        float hv = sigmoidf_(go) * tanhf(c2);
        hS[t] = hv;
        qstar[t] = hv;  // q-part of q_star
    }
}

__global__ void s2s_scores(const float* __restrict__ xh, const float* __restrict__ hS,
                           float* __restrict__ sbuf, unsigned* __restrict__ smax, int N) {
    int gid = blockIdx.x * blockDim.x + threadIdx.x;
    int n = gid >> 5, lane = gid & 31;
    if (n >= N) return;
    const float* row = xh + (size_t)n * 128;
    float s = 0;
#pragma unroll
    for (int r = 0; r < 4; ++r) s += row[r * 32 + lane] * hS[r * 32 + lane];
#pragma unroll
    for (int off = 16; off; off >>= 1) s += __shfl_xor(s, off, 32);
    if (lane == 0) { sbuf[n] = s; atomicMax(smax, fenc(s)); }
}

__global__ void s2s_exp(float* __restrict__ sbuf, const unsigned* __restrict__ smax,
                        float* __restrict__ ssum, int N) {
    int i = blockIdx.x * blockDim.x + threadIdx.x;
    if (i >= N) return;
    float p = __expf(sbuf[i] - fdec(*smax));
    sbuf[i] = p;
    atomicAdd(ssum, p);
}

// r += (p/S) * xh ; 128-thread block covers 64 nodes, 1 atomic per column/block
__global__ void s2s_r(const float* __restrict__ xh, const float* __restrict__ sbuf,
                      const float* __restrict__ ssum, float* __restrict__ r, int N) {
    int c = threadIdx.x;
    int n0 = blockIdx.x * 64;
    int n1 = n0 + 64 < N ? n0 + 64 : N;
    float inv = 1.0f / (*ssum);
    float acc = 0;
    for (int n = n0; n < n1; ++n) acc += sbuf[n] * xh[(size_t)n * 128 + c];
    atomicAdd(&r[c], acc * inv);
}

__global__ void head_mlp(const float* __restrict__ qstar, const float* __restrict__ v,
                         const float* __restrict__ w1, const float* __restrict__ b1,
                         const float* __restrict__ w2, const float* __restrict__ b2,
                         const float* __restrict__ w3, const float* __restrict__ b3,
                         float* __restrict__ out) {
    __shared__ float comb[384];
    __shared__ float o1[128];
    __shared__ float o2[64];
    int t = threadIdx.x;  // 256
    if (t < 256) comb[t] = qstar[t];
    if (t < 128) comb[256 + t] = v[t];
    __syncthreads();
    if (t < 128) {
        float s = b1[t];
        for (int k = 0; k < 384; ++k) s += comb[k] * w1[(size_t)k * 128 + t];
        o1[t] = s > 0.0f ? s : 0.0f;
    }
    __syncthreads();
    if (t < 64) {
        float s = b2[t];
        for (int k = 0; k < 128; ++k) s += o1[k] * w2[(size_t)k * 64 + t];
        o2[t] = s > 0.0f ? s : 0.0f;
    }
    __syncthreads();
    float s = b3[t];
    for (int k = 0; k < 64; ++k) s += o2[k] * w3[(size_t)k * 256 + t];
    out[t] = s > 0.0f ? s : 0.0f;
}

// ============================================================================
extern "C" void kernel_launch(void* const* d_in, const int* in_sizes, int n_in,
                              void* d_out, int out_size, void* d_ws, size_t ws_size,
                              hipStream_t stream) {
    const float* x        = (const float*)d_in[0];
    const int*   eidx     = (const int*)d_in[1];
    const float* eattr    = (const float*)d_in[2];
    const float* vnf      = (const float*)d_in[3];
    const float* w_node   = (const float*)d_in[4];
    const float* b_node   = (const float*)d_in[5];
    const float* w_eemb   = (const float*)d_in[6];
    const float* b_eemb   = (const float*)d_in[7];
    const float* w_eproj  = (const float*)d_in[8];
    const float* b_eproj  = (const float*)d_in[9];
    const float* w_vnf    = (const float*)d_in[10];
    const float* b_vnf    = (const float*)d_in[11];
    const float* attn_w1  = (const float*)d_in[12];
    const float* attn_b1  = (const float*)d_in[13];
    const float* attn_w2  = (const float*)d_in[14];
    const float* attn_b2  = (const float*)d_in[15];
    const float* attn_w3  = (const float*)d_in[16];
    const float* attn_b3  = (const float*)d_in[17];
    const float* gat_w    = (const float*)d_in[18];
    const float* att_src  = (const float*)d_in[19];
    const float* att_dst  = (const float*)d_in[20];
    const float* gat_we   = (const float*)d_in[21];
    const float* att_e    = (const float*)d_in[22];
    const float* gat_b    = (const float*)d_in[23];
    const float* ln_g     = (const float*)d_in[24];
    const float* ln_b     = (const float*)d_in[25];
    const float* venc_w1  = (const float*)d_in[26];
    const float* venc_b1  = (const float*)d_in[27];
    const float* venc_w2  = (const float*)d_in[28];
    const float* venc_b2  = (const float*)d_in[29];
    const float* lstm_wih = (const float*)d_in[30];
    const float* lstm_whh = (const float*)d_in[31];
    const float* lstm_bih = (const float*)d_in[32];
    const float* lstm_bhh = (const float*)d_in[33];
    const float* out_w1   = (const float*)d_in[34];
    const float* out_b1   = (const float*)d_in[35];
    const float* out_w2   = (const float*)d_in[36];
    const float* out_b2   = (const float*)d_in[37];
    const float* out_w3   = (const float*)d_in[38];
    const float* out_b3   = (const float*)d_in[39];
    float* out = (float*)d_out;

    const int N = in_sizes[0] / 8;
    const int E = in_sizes[2] / 4;
    const int* src = eidx;
    const int* dst = eidx + E;

    // --- bump allocator over workspace (regions reused across phases) -------
    char* ws = (char*)d_ws;
    size_t off = 0;
    auto alloc = [&](size_t bytes) -> void* {
        void* p = ws + off;
        off = (off + bytes + 255) & ~(size_t)255;
        return p;
    };
    size_t bigE = (size_t)E * 512;                 // E*128 f32 bytes
    size_t bigN = (size_t)N * 2048;                // N*512 f32 bytes
    size_t big  = bigE > bigN ? bigE : bigN;       // 102.4 MB
    float*    R1     = (float*)alloc(big);         // ea -> h1 -> xs
    float*    R2     = (float*)alloc(big);         // ef -> agg
    float*    e_enh  = (float*)alloc(bigE);        // persists through layers
    _Float16* R4     = (_Float16*)alloc(bigE / 2); // f16 staging (ea/ef/h1/xh)
    float*    h2buf  = (float*)alloc((size_t)E * 64 * 4);
    float*    xhA    = (float*)alloc((size_t)N * 512);
    float*    xhB    = (float*)alloc((size_t)N * 512);
    float*    attbuf = (float*)alloc((size_t)E * 4);
    float*    escore = (float*)alloc((size_t)E * 16);
    float*    abuf   = (float*)alloc((size_t)E * 16);
    float*    asrc   = (float*)alloc((size_t)N * 16);
    float*    adst   = (float*)alloc((size_t)N * 16);
    unsigned* amax   = (unsigned*)alloc((size_t)N * 16);
    float*    denom  = (float*)alloc((size_t)N * 16);
    float*    weff   = (float*)alloc(2048);
    _Float16* wep_p  = (_Float16*)alloc(128 * 128 * 2);
    _Float16* w1a_p  = (_Float16*)alloc(128 * 128 * 2);
    _Float16* w2_p   = (_Float16*)alloc(128 * 64 * 2);
    _Float16* gwp[3];
    for (int l = 0; l < 3; ++l) gwp[l] = (_Float16*)alloc(128 * 512 * 2);
    float*    bias1  = (float*)alloc(512);
    float*    vvec   = (float*)alloc(512);
    float*    qstar  = (float*)alloc(1024);
    float*    hS     = (float*)alloc(512);
    float*    cS     = (float*)alloc(512);
    unsigned* smax   = (unsigned*)alloc(256);
    float*    ssum   = (float*)alloc(256);
    float*    sbuf   = (float*)alloc((size_t)N * 4);
    (void)n_in; (void)out_size; (void)ws_size;

    // --- weight prep --------------------------------------------------------
    pack_b_f16<<<CDIV(128 * 128, 256), 256, 0, stream>>>(w_eproj, wep_p, 128, 128);
    pack_b_f16<<<CDIV(128 * 128, 256), 256, 0, stream>>>(attn_w1, w1a_p, 128, 128); // top half
    pack_b_f16<<<CDIV(128 * 64, 256), 256, 0, stream>>>(attn_w2, w2_p, 128, 64);
    for (int l = 0; l < 3; ++l)
        pack_b_f16<<<CDIV(128 * 512, 256), 256, 0, stream>>>(gat_w + (size_t)l * 128 * 512,
                                                             gwp[l], 128, 512);
    make_bias1<<<1, 128, 0, stream>>>(vnf, w_vnf, b_vnf, attn_w1, attn_b1, bias1);

    // --- embeddings ---------------------------------------------------------
    node_embed<<<N, 128, 0, stream>>>(x, w_node, b_node, xhA);
    edge_embed<<<E, 128, 0, stream>>>(eattr, w_eemb, b_eemb, R1);      // ea

    // --- edge projection + attention MLP (WMMA GEMMs) -----------------------
    const int Etiles = E / 16, Ntiles = N / 16;
    cvt_f32_f16<<<2048, 256, 0, stream>>>(R1, R4, (size_t)E * 128);    // ea -> f16
    gemm_wmma_f16<<<dim3(CDIV(Etiles, 8), 2), 256, 0, stream>>>(R4, wep_p, R2, b_eproj,
                                                                E, 128, 128, 0);  // ef
    cvt_f32_f16<<<2048, 256, 0, stream>>>(R2, R4, (size_t)E * 128);    // ef -> f16
    gemm_wmma_f16<<<dim3(CDIV(Etiles, 8), 2), 256, 0, stream>>>(R4, w1a_p, R1, bias1,
                                                                E, 128, 128, 1);  // h1
    cvt_f32_f16<<<2048, 256, 0, stream>>>(R1, R4, (size_t)E * 128);    // h1 -> f16
    gemm_wmma_f16<<<dim3(CDIV(Etiles, 8), 1), 256, 0, stream>>>(R4, w2_p, h2buf, attn_b2,
                                                                E, 128, 64, 1);   // h2
    edge_att<<<CDIV(E * 32, 256), 256, 0, stream>>>(h2buf, attn_w3, attn_b3, R2,
                                                    e_enh, attbuf, E);

    // --- 3 GAT layers -------------------------------------------------------
    float* xh_cur = xhA;
    float* xh_nxt = xhB;
    for (int l = 0; l < 3; ++l) {
        cvt_f32_f16<<<2048, 256, 0, stream>>>(xh_cur, R4, (size_t)N * 128);
        gemm_wmma_f16<<<dim3(CDIV(Ntiles, 8), 8), 256, 0, stream>>>(R4, gwp[l], R1, nullptr,
                                                                    N, 128, 512, 0); // xs
        node_scores<<<CDIV(N * 32, 256), 256, 0, stream>>>(R1, att_src + (size_t)l * 512,
                                                           att_dst + (size_t)l * 512,
                                                           asrc, adst, N);
        make_weff<<<2, 256, 0, stream>>>(gat_we + (size_t)l * 128 * 512,
                                         att_e + (size_t)l * 512, weff);
        edge_score<<<CDIV(E * 32, 256), 256, 0, stream>>>(e_enh, weff, escore, E);
        hipMemsetAsync(amax, 0, (size_t)N * 16, stream);
        hipMemsetAsync(denom, 0, (size_t)N * 16, stream);
        hipMemsetAsync(R2, 0, (size_t)N * 2048, stream);               // agg
        edge_logits<<<CDIV(E * 4, 256), 256, 0, stream>>>(src, dst, asrc, adst, escore,
                                                          abuf, amax, E * 4);
        edge_exp<<<CDIV(E * 4, 256), 256, 0, stream>>>(dst, abuf, amax, denom, E * 4);
        edge_scatter<<<E, 128, 0, stream>>>(src, dst, abuf, denom, R1, R2);
        node_update<<<N, 128, 0, stream>>>(R2, gat_b + (size_t)l * 128,
                                           ln_g + (size_t)l * 128, ln_b + (size_t)l * 128,
                                           xh_cur, xh_nxt);
        float* t = xh_cur; xh_cur = xh_nxt; xh_nxt = t;
    }

    // --- vnf encoder + Set2Set (3 steps) + head MLP -------------------------
    venc_kernel<<<1, 128, 0, stream>>>(vnf, venc_w1, venc_b1, venc_w2, venc_b2, vvec);
    hipMemsetAsync(qstar, 0, 1024, stream);
    hipMemsetAsync(hS, 0, 512, stream);
    hipMemsetAsync(cS, 0, 512, stream);
    for (int step = 0; step < 3; ++step) {
        lstm_step<<<1, 512, 0, stream>>>(lstm_wih, lstm_whh, lstm_bih, lstm_bhh,
                                         qstar, hS, cS);
        hipMemsetAsync(smax, 0, 4, stream);
        hipMemsetAsync(ssum, 0, 4, stream);
        hipMemsetAsync(qstar + 128, 0, 512, stream);
        s2s_scores<<<CDIV(N * 32, 256), 256, 0, stream>>>(xh_cur, hS, sbuf, smax, N);
        s2s_exp<<<CDIV(N, 256), 256, 0, stream>>>(sbuf, smax, ssum, N);
        s2s_r<<<CDIV(N, 64), 128, 0, stream>>>(xh_cur, sbuf, ssum, qstar + 128, N);
    }
    head_mlp<<<1, 256, 0, stream>>>(qstar, vvec, out_w1, out_b1, out_w2, out_b2,
                                    out_w3, out_b3, out);
}
